// Model_54013508714831
// MI455X (gfx1250) — matrix operand order repaired and checked by reading, not compiled, not run on hardware
//
#include <hip/hip_runtime.h>
#include <hip/hip_bf16.h>

// ---------------- problem constants ----------------
#define TS   512
#define BS   128
#define INS  128
#define HD   512
#define OD   64

#define NWG        16                 // workgroups in persistent RNN kernel
#define SLICE      32                 // HD / NWG columns owned per WG
#define RNN_THREADS 512               // 16 wave32s -> one 16x16 tile each
#define OUT_THREADS 256               // 8 wave32s

#define DECAY   0.8187307530779818f   // exp(-0.2)
#define NSCALE  0.2870888412830514f   // sqrt(1-exp(-0.4)) * 0.5
#define AN      0.1f

#define INP (INS + 8)                 // padded K strides (f16 elems) for LDS
#define HP  (HD  + 8)

typedef __attribute__((ext_vector_type(16))) _Float16 v16h;
typedef __attribute__((ext_vector_type(8)))  _Float16 v8h;
typedef __attribute__((ext_vector_type(8)))  float    v8f;

// ---------------- WMMA helpers ----------------
__device__ __forceinline__ v8f wmma16(v16h a, v16h b, v8f c) {
  // v_wmma_f32_16x16x32_f16
  return __builtin_amdgcn_wmma_f32_16x16x32_f16(false, a, false, b, (short)0, c,
                                                false, false);
}

__device__ __forceinline__ v16h pack16(const _Float16* p0, const _Float16* p1) {
  v8h lo = *(const v8h*)p0;   // 16B load
  v8h hi = *(const v8h*)p1;   // 16B load
  return __builtin_shufflevector(lo, hi, 0,1,2,3,4,5,6,7,8,9,10,11,12,13,14,15);
}

// A fragment (16x32 f16, row = batch), k = k0 + lane offset (0/8); lane half h reads
// a[0..15] = row[k0+16h .. k0+16h+15]
__device__ __forceinline__ v16h load_a(const _Float16* rowp, int k) {
  const int kk = (k & ~31) + 2 * (k & 31);
  return pack16(rowp + kk, rowp + kk + 8);
}
// B fragment (32x16 f16) from LDS transposed-weight row (one output column),
// k already includes the lane offset (0/16): 16 contiguous f16.
__device__ __forceinline__ v16h load_b(const _Float16* rowp, int k) {
  return pack16(rowp + k, rowp + k + 8);
}

// ---------------- device-wide barrier (monotonic counter) ----------------
__device__ __forceinline__ void grid_barrier(unsigned* bar, unsigned phase) {
  __threadfence();                     // make this WG's h stores device-visible
  __syncthreads();
  if (threadIdx.x == 0) {
    __hip_atomic_fetch_add(bar, 1u, __ATOMIC_RELEASE, __HIP_MEMORY_SCOPE_AGENT);
    const unsigned target = phase * NWG;
    while (__hip_atomic_load(bar, __ATOMIC_ACQUIRE, __HIP_MEMORY_SCOPE_AGENT) < target)
      __builtin_amdgcn_s_sleep(2);
  }
  __syncthreads();
  __threadfence();                     // acquire: don't read stale h
}

// ---------------- transposed weight slice -> LDS (f32 global -> f16 LDS) ----------
__device__ __forceinline__ void load_wt_slice(_Float16* dst, const float* w,
                                              int K, int Kp, int nBaseG) {
  // w is [K, HD] row-major; dst is [SLICE][Kp] (dst[n][k] = w[k][nBaseG+n])
  for (int idx = threadIdx.x; idx < K * SLICE; idx += RNN_THREADS) {
    int n = idx & (SLICE - 1);
    int k = idx >> 5;
    dst[n * Kp + k] = (_Float16)w[(size_t)k * HD + nBaseG + n];
  }
}

// ---------------- init: reset barrier + zero h ping-pong buffers ----------------
__global__ void rnn_init_kernel(unsigned* bar, _Float16* h1, _Float16* h2) {
  size_t i = (size_t)blockIdx.x * blockDim.x + threadIdx.x;
  if (i == 0) *bar = 0u;
  if (i < (size_t)2 * BS * HD) {
    h1[i] = (_Float16)0.0f;
    h2[i] = (_Float16)0.0f;
  }
}

// ---------------- x f32 -> f16 ----------------
__global__ void cvt_kernel(const float* __restrict__ x1, const float* __restrict__ x2,
                           _Float16* __restrict__ x1h, _Float16* __restrict__ x2h,
                           int n) {
  int i = blockIdx.x * blockDim.x + threadIdx.x;
  if (i < n) {
    x1h[i] = (_Float16)x1[i];
    x2h[i] = (_Float16)x2[i];
  }
}

// ---------------- persistent recurrence kernel ----------------
__global__ __launch_bounds__(RNN_THREADS, 1)
void rnn_kernel(const _Float16* __restrict__ x1h, const _Float16* __restrict__ x2h,
                const float* __restrict__ eps1, const float* __restrict__ eps2,
                const float* __restrict__ w_in1, const float* __restrict__ w_in2,
                const float* __restrict__ w_r11, const float* __restrict__ w_r12,
                const float* __restrict__ w_r21, const float* __restrict__ w_r22,
                _Float16* __restrict__ h1buf, _Float16* __restrict__ h2buf,
                _Float16* __restrict__ hs, unsigned* bar) {
  extern __shared__ char smem_raw[];
  _Float16* lds = (_Float16*)smem_raw;
  _Float16* Lin1 = lds;                       // [SLICE][INP]
  _Float16* Lr11 = Lin1 + SLICE * INP;        // [SLICE][HP]
  _Float16* Lr21 = Lr11 + SLICE * HP;
  _Float16* Lin2 = Lr21 + SLICE * HP;
  _Float16* Lr12 = Lin2 + SLICE * INP;
  _Float16* Lr22 = Lr12 + SLICE * HP;

  const int wg     = blockIdx.x;              // 0..NWG-1
  const int nBaseG = wg * SLICE;              // global hidden-column base

  // one-time weight slice load (transposed, f16, padded rows)
  load_wt_slice(Lin1, w_in1, INS, INP, nBaseG);
  load_wt_slice(Lr11, w_r11, HD,  HP,  nBaseG);
  load_wt_slice(Lr21, w_r21, HD,  HP,  nBaseG);
  load_wt_slice(Lin2, w_in2, INS, INP, nBaseG);
  load_wt_slice(Lr12, w_r12, HD,  HP,  nBaseG);
  load_wt_slice(Lr22, w_r22, HD,  HP,  nBaseG);
  __syncthreads();

  const int wave  = threadIdx.x >> 5;
  const int lane  = threadIdx.x & 31;
  const int mBase = (wave >> 1) * 16;         // 8 batch bands of 16
  const int nTile = (wave & 1) * 16;          // 2 column tiles of 16 in slice
  const int lhalf = lane >> 4;                // 0 / 1
  const int l15   = lane & 15;
  const int hoff  = lhalf * 8;                // A k-offset and C row offset
  const int koffB = lhalf * 16;               // B k-offset
  const int aRow  = mBase + l15;              // A-fragment source row (batch)
  const int colG  = nBaseG + nTile + l15;     // this lane's output hidden column
  const _Float16* Bin1 = Lin1 + (nTile + l15) * INP;
  const _Float16* Br11 = Lr11 + (nTile + l15) * HP;
  const _Float16* Br21 = Lr21 + (nTile + l15) * HP;
  const _Float16* Bin2 = Lin2 + (nTile + l15) * INP;
  const _Float16* Br12 = Lr12 + (nTile + l15) * HP;
  const _Float16* Br22 = Lr22 + (nTile + l15) * HP;

  // persistent per-thread state, laid out exactly as the WMMA C fragment:
  // element r lives at batch row (mBase + r + 8*lhalf), hidden column colG
  float h1r[8], h2r[8], n1r[8], n2r[8];
#pragma unroll
  for (int r = 0; r < 8; ++r) { h1r[r] = 0.f; h2r[r] = 0.f; n1r[r] = 0.f; n2r[r] = 0.f; }

  unsigned phase = 0;
  for (int t = 0; t < TS; ++t) {
    const int p  = t & 1;
    const int rp = p ^ 1;
    const _Float16* h1rd = h1buf + (size_t)rp * BS * HD;
    _Float16*       h1wr = h1buf + (size_t)p  * BS * HD;
    const _Float16* h2rd = h2buf + (size_t)rp * BS * HD;
    _Float16*       h2wr = h2buf + (size_t)p  * BS * HD;

    // ================= phase 1: h1_new =================
    v8f acc = {};
#pragma unroll
    for (int r = 0; r < 8; ++r) {               // OU noise update; acc starts at n1
      const int row = mBase + r + hoff;
      n1r[r] = n1r[r] * DECAY +
               NSCALE * eps1[(size_t)t * BS * HD + (size_t)row * HD + colG];
      acc[r] = n1r[r];
    }
    {
      const _Float16* Ar = x1h + (size_t)t * BS * INS + (size_t)aRow * INS;
#pragma unroll
      for (int k = 0; k < INS; k += 32)
        acc = wmma16(load_a(Ar, k + hoff), load_b(Bin1, k + koffB), acc);
    }
    {
      const _Float16* Ar = h1rd + (size_t)aRow * HD;
#pragma unroll 4
      for (int k = 0; k < HD; k += 32)
        acc = wmma16(load_a(Ar, k + hoff), load_b(Br11, k + koffB), acc);
    }
    {
      const _Float16* Ar = h2rd + (size_t)aRow * HD;
#pragma unroll 4
      for (int k = 0; k < HD; k += 32)
        acc = wmma16(load_a(Ar, k + hoff), load_b(Br21, k + koffB), acc);
    }
#pragma unroll
    for (int r = 0; r < 8; ++r) {
      const float sg = 1.0f / (1.0f + __expf(-acc[r]));
      h1r[r] = h1r[r] * (1.0f - AN) + AN * sg;
      const int row = mBase + r + hoff;
      h1wr[(size_t)row * HD + colG] = (_Float16)h1r[r];
    }
    grid_barrier(bar, ++phase);                 // all WGs see full h1_new

    // ================= phase 2: h2_new (uses NEW h1) =================
    acc = v8f{};
#pragma unroll
    for (int r = 0; r < 8; ++r) {
      const int row = mBase + r + hoff;
      n2r[r] = n2r[r] * DECAY +
               NSCALE * eps2[(size_t)t * BS * HD + (size_t)row * HD + colG];
      acc[r] = n2r[r];
    }
    {
      const _Float16* Ar = x2h + (size_t)t * BS * INS + (size_t)aRow * INS;
#pragma unroll
      for (int k = 0; k < INS; k += 32)
        acc = wmma16(load_a(Ar, k + hoff), load_b(Bin2, k + koffB), acc);
    }
    {
      const _Float16* Ar = h1wr + (size_t)aRow * HD;   // NEW h1
#pragma unroll 4
      for (int k = 0; k < HD; k += 32)
        acc = wmma16(load_a(Ar, k + hoff), load_b(Br12, k + koffB), acc);
    }
    {
      const _Float16* Ar = h2rd + (size_t)aRow * HD;   // old h2
#pragma unroll 4
      for (int k = 0; k < HD; k += 32)
        acc = wmma16(load_a(Ar, k + hoff), load_b(Br22, k + koffB), acc);
    }
#pragma unroll
    for (int r = 0; r < 8; ++r) {
      const float sg = 1.0f / (1.0f + __expf(-acc[r]));
      h2r[r] = h2r[r] * (1.0f - AN) + AN * sg;
      const int row = mBase + r + hoff;
      h2wr[(size_t)row * HD + colG] = (_Float16)h2r[r];
      hs[(size_t)t * BS * HD + (size_t)row * HD + colG] = (_Float16)(h1r[r] + h2r[r]);
    }
    grid_barrier(bar, ++phase);                 // next step may ping-pong
  }
}

// ---------------- output projection: y = hs[T*B,H] @ w_out[H,O] (x2) -------------
__global__ __launch_bounds__(OUT_THREADS, 2)
void out_kernel(const _Float16* __restrict__ hs,
                const float* __restrict__ wdm, const float* __restrict__ wem,
                float* __restrict__ out) {
  extern __shared__ char smem_raw[];
  _Float16* Ldm = (_Float16*)smem_raw;          // [OD][HP]
  _Float16* Lem = Ldm + OD * HP;
  for (int idx = threadIdx.x; idx < HD * OD; idx += OUT_THREADS) {
    int n = idx & (OD - 1);
    int k = idx >> 6;
    Ldm[n * HP + k] = (_Float16)wdm[(size_t)k * OD + n];
    Lem[n * HP + k] = (_Float16)wem[(size_t)k * OD + n];
  }
  __syncthreads();

  const int wave  = threadIdx.x >> 5;
  const int lane  = threadIdx.x & 31;
  const int lhalf = lane >> 4;
  const int l15   = lane & 15;
  const int hoff  = lhalf * 8;
  const int koffB = lhalf * 16;
  const size_t mBase = (size_t)blockIdx.x * 128 + (size_t)wave * 16;
  const _Float16* Ar = hs + (mBase + l15) * HD;

  v8f accd[4] = {}, acce[4] = {};
  for (int k = 0; k < HD; k += 32) {
    v16h a = load_a(Ar, k + hoff);
#pragma unroll
    for (int j = 0; j < 4; ++j) {
      const int nrow = j * 16 + l15;
      accd[j] = wmma16(a, load_b(Ldm + nrow * HP, k + koffB), accd[j]);
      acce[j] = wmma16(a, load_b(Lem + nrow * HP, k + koffB), acce[j]);
    }
  }
  const size_t emOff = (size_t)TS * BS * OD;
#pragma unroll
  for (int j = 0; j < 4; ++j)
#pragma unroll
    for (int r = 0; r < 8; ++r) {
      const size_t row = mBase + r + hoff;      // flat (t*B + b)
      const int    col = j * 16 + l15;
      out[row * OD + col]         = accd[j][r];
      out[emOff + row * OD + col] = acce[j][r];
    }
}

// ---------------- host entry ----------------
extern "C" void kernel_launch(void* const* d_in, const int* in_sizes, int n_in,
                              void* d_out, int out_size, void* d_ws, size_t ws_size,
                              hipStream_t stream) {
  (void)in_sizes; (void)n_in; (void)out_size; (void)ws_size;

  const float* x1    = (const float*)d_in[0];
  const float* x2    = (const float*)d_in[1];
  const float* eps1  = (const float*)d_in[2];
  const float* eps2  = (const float*)d_in[3];
  const float* w_in1 = (const float*)d_in[4];
  const float* w_in2 = (const float*)d_in[5];
  const float* w_r11 = (const float*)d_in[6];
  const float* w_r12 = (const float*)d_in[7];
  const float* w_r21 = (const float*)d_in[8];
  const float* w_r22 = (const float*)d_in[9];
  const float* w_dm  = (const float*)d_in[10];
  const float* w_em  = (const float*)d_in[11];
  float* out = (float*)d_out;

  // workspace layout (bytes)
  char* ws = (char*)d_ws;
  const size_t XH_SZ = (size_t)TS * BS * INS * sizeof(_Float16);   // 16 MB
  const size_t HB_SZ = (size_t)2 * BS * HD * sizeof(_Float16);     // 256 KB (ping-pong)
  unsigned*  bar = (unsigned*)ws;
  _Float16*  x1h = (_Float16*)(ws + 256);
  _Float16*  x2h = (_Float16*)(ws + 256 + XH_SZ);
  _Float16*  h1b = (_Float16*)(ws + 256 + 2 * XH_SZ);
  _Float16*  h2b = (_Float16*)(ws + 256 + 2 * XH_SZ + HB_SZ);
  _Float16*  hsb = (_Float16*)(ws + 256 + 2 * XH_SZ + 2 * HB_SZ);  // [T,B,H] f16

  // 1) reset barrier + zero hidden-state ping-pong buffers (replay-safe)
  {
    const int n = 2 * BS * HD;
    rnn_init_kernel<<<(n + 255) / 256, 256, 0, stream>>>(bar, h1b, h2b);
  }
  // 2) convert x1,x2 to f16
  {
    const int n = TS * BS * INS;
    cvt_kernel<<<(n + 255) / 256, 256, 0, stream>>>(x1, x2, x1h, x2h, n);
  }
  // 3) persistent recurrence: 16 WGs, weights resident in LDS
  {
    const size_t lds_bytes =
        (size_t)(2 * (SLICE * INP + 2 * SLICE * HP)) * sizeof(_Float16); // ~147 KB
    rnn_kernel<<<NWG, RNN_THREADS, lds_bytes, stream>>>(
        x1h, x2h, eps1, eps2, w_in1, w_in2, w_r11, w_r12, w_r21, w_r22,
        h1b, h2b, hsb, bar);
  }
  // 4) output projection GEMM
  {
    const size_t lds_bytes = (size_t)(2 * OD * HP) * sizeof(_Float16);   // ~130 KB
    const int nblk = (TS * BS) / 128;                                    // 512
    out_kernel<<<nblk, OUT_THREADS, lds_bytes, stream>>>(hsb, w_dm, w_em, out);
  }
}
